// MoeLayer_ddp_86620900426404
// MI455X (gfx1250) — matrix operand order repaired and checked by reading, not compiled, not run on hardware
//
#include <hip/hip_runtime.h>
#include <stdint.h>

// ---------------------------------------------------------------------------
// MoE top-2 layer for MI455X (gfx1250): bf16 WMMA FFN with fused relu
// B=1, T=2048, D=1024, E=8, DFF=4096, capacity = 2048 (no drops possible)
// ---------------------------------------------------------------------------

typedef __attribute__((ext_vector_type(16))) __bf16 v16bf;
typedef __attribute__((ext_vector_type(8)))  __bf16 v8bf;
typedef __attribute__((ext_vector_type(8)))  float  v8f;

#define S_TOK  2048
#define DMODEL 1024
#define NEXP   8
#define DFF_   4096
#define CAP    2048
#define MROWS  32          // rows (expert slots) per FFN block

#define XS_STRIDE 1040     // 1024 + 16 halves pad -> row stride 2080B (8-bank skew)
#define HS_STRIDE 136      // 128 + 8 halves pad   -> row stride 272B  (4-bank skew)

// dynamic LDS carve-up (bytes)
#define XS_BYTES   (MROWS * XS_STRIDE * 2)              // 66560
#define HS_BYTES   (MROWS * HS_STRIDE * 2)              // 8704
#define TOK_OFF    (XS_BYTES + HS_BYTES)                // 75264
#define SMEM_BYTES (TOK_OFF + MROWS * 4 + MROWS * 4)    // + tok[32] + gate[32]

__device__ __forceinline__ unsigned short f2bf_bits(float f) {
  unsigned int u = __builtin_bit_cast(unsigned int, f);
  unsigned int r = u + 0x7FFFu + ((u >> 16) & 1u);  // round-to-nearest-even
  return (unsigned short)(r >> 16);
}
__device__ __forceinline__ __bf16 f2bf(float f) {
  unsigned short h = f2bf_bits(f);
  return __builtin_bit_cast(__bf16, h);
}

// A fragment (16x32 bf16, ISA 7.12.2): lane%16 = row M, hi = lane>>4.
// half j (0..15): K = (j&7) + (j>=8 ? 16 : 0) + 8*hi  -> two contiguous
// 8-half (16B) groups per lane => two b128 loads from row-major storage.
__device__ __forceinline__ v16bf load_a_frag(const __bf16* row_base, int k0, int hi) {
  const __bf16* p = row_base + k0 + hi * 8;
  v8bf lo = *(const v8bf*)(p);
  v8bf hh = *(const v8bf*)(p + 16);
  return __builtin_shufflevector(lo, hh, 0,1,2,3,4,5,6,7,8,9,10,11,12,13,14,15);
}

// B fragment (32x16 bf16): pre-packed so each lane's 16 halves are contiguous.
__device__ __forceinline__ v16bf load_b_frag(const __bf16* p) {
  v8bf lo = *(const v8bf*)(p);
  v8bf hh = *(const v8bf*)(p + 8);
  return __builtin_shufflevector(lo, hh, 0,1,2,3,4,5,6,7,8,9,10,11,12,13,14,15);
}

// ---------------------------------------------------------------------------
// 0) zero output (atomically accumulated later)
// ---------------------------------------------------------------------------
__global__ __launch_bounds__(256) void zero_kernel(float* __restrict__ p, int n) {
  int i = blockIdx.x * 256 + threadIdx.x;
  if (i < n) p[i] = 0.0f;
}

// ---------------------------------------------------------------------------
// 1) pack fp32 weight [K, ncols] into bf16 WMMA B-fragment tiles, staged
// through LDS so both global reads and writes are fully coalesced.
// Block = one K-band of 32 rows x 256 columns (16 tiles).
// tile id = kt * (ncols/16) + nt; tile = 32 lanes x 16 halves (lane-major);
// lane holds column nt*16+(lane&15); half j holds K = kt*32 + (j&7) + 8*(lane>>4) + 16*(j>>3).
// ---------------------------------------------------------------------------
__global__ __launch_bounds__(256) void pack_b_kernel(const float* __restrict__ w,
                                                     unsigned short* __restrict__ packed,
                                                     int ncols) {
  __shared__ unsigned short lds[32][264];   // 256 cols + 8 pad
  int ngrp = ncols >> 8;                    // column groups of 256
  int kt   = blockIdx.x / ngrp;
  int g    = blockIdx.x % ngrp;
  int col0 = g << 8;
  int tid  = threadIdx.x;
  // coalesced read: 32 rows x 256 cols
  for (int o = tid; o < 32 * 256; o += 256) {
    int r = o >> 8, c = o & 255;
    lds[r][c] = f2bf_bits(w[(size_t)(kt * 32 + r) * ncols + col0 + c]);
  }
  __syncthreads();
  // coalesced write: 16 contiguous tiles, fragment order (lane*16 + j)
  size_t base = ((size_t)kt * (ncols >> 4) + (size_t)g * 16) * 512;
  for (int o = tid; o < 16 * 512; o += 256) {
    int nt_local = o >> 9;
    int within   = o & 511;                 // = lane*16 + j
    int lane     = within >> 4;
    int j        = within & 15;
    int Kl       = (j & 7) + 8 * (lane >> 4) + ((j >> 3) << 4);
    int c        = nt_local * 16 + (lane & 15);
    packed[base + o] = lds[Kl][c];
  }
}

// ---------------------------------------------------------------------------
// 2) gating: one wave per token. logits = x[s]·Wg, softmax, top-2, normalize.
// ---------------------------------------------------------------------------
__global__ __launch_bounds__(256) void gate_kernel(const float* __restrict__ x,
                                                   const float* __restrict__ Wg,
                                                   int* __restrict__ e1, int* __restrict__ e2,
                                                   float* __restrict__ g1, float* __restrict__ g2) {
  int s    = blockIdx.x * 8 + (threadIdx.x >> 5);
  int lane = threadIdx.x & 31;
  float acc[NEXP] = {0.f,0.f,0.f,0.f,0.f,0.f,0.f,0.f};
  const float* xr = x + (size_t)s * DMODEL;
  for (int d = lane; d < DMODEL; d += 32) {
    float xv = xr[d];
    const float* wr = Wg + (size_t)d * NEXP;
#pragma unroll
    for (int e = 0; e < NEXP; ++e) acc[e] += xv * wr[e];
  }
#pragma unroll
  for (int e = 0; e < NEXP; ++e) {
    for (int off = 16; off > 0; off >>= 1) acc[e] += __shfl_xor(acc[e], off, 32);
  }
  if (lane == 0) {
    float lmax = acc[0];
#pragma unroll
    for (int e = 1; e < NEXP; ++e) lmax = fmaxf(lmax, acc[e]);
    float ex[NEXP], sum = 0.f;
#pragma unroll
    for (int e = 0; e < NEXP; ++e) { ex[e] = __expf(acc[e] - lmax); sum += ex[e]; }
    int i1 = 0; float b1 = acc[0];
#pragma unroll
    for (int e = 1; e < NEXP; ++e) if (acc[e] > b1) { b1 = acc[e]; i1 = e; }
    int i2 = -1; float b2 = -3.4e38f;
#pragma unroll
    for (int e = 0; e < NEXP; ++e) if (e != i1 && acc[e] > b2) { b2 = acc[e]; i2 = e; }
    float gg1 = ex[i1] / sum, gg2 = ex[i2] / sum;
    float denom = fmaxf(gg1 + gg2, 1.1920929e-7f);
    e1[s] = i1; e2[s] = i2;
    g1[s] = gg1 / denom; g2[s] = gg2 / denom;
  }
}

// ---------------------------------------------------------------------------
// 3) parallel slot scan (single block, 256 threads; 8 tokens/thread).
// Per-thread per-expert histogram -> cross-thread exclusive prefix (per
// expert) -> replay to emit compacted (token, gate) lists. Matches the
// reference cumsum ordering: pass1 slots [0,count1), pass2 [count1, total).
// ---------------------------------------------------------------------------
__global__ __launch_bounds__(256) void scan_kernel(const int* __restrict__ e1,
                                                   const int* __restrict__ e2,
                                                   const float* __restrict__ g1,
                                                   const float* __restrict__ g2,
                                                   int* __restrict__ token_of,
                                                   float* __restrict__ gate_of,
                                                   int* __restrict__ total) {
  __shared__ int hist[256 * NEXP];
  __shared__ int tot1[NEXP];
  int t  = threadIdx.x;
  int s0 = t * 8;
  int le[8], h[NEXP];

  // ---- pass 1: first-choice experts ----
#pragma unroll
  for (int e = 0; e < NEXP; ++e) h[e] = 0;
#pragma unroll
  for (int i = 0; i < 8; ++i) { le[i] = e1[s0 + i]; h[le[i]]++; }
#pragma unroll
  for (int e = 0; e < NEXP; ++e) hist[t * NEXP + e] = h[e];
  __syncthreads();
  if (t < NEXP) {             // serial exclusive scan over 256 entries (LDS)
    int run = 0;
    for (int i = 0; i < 256; ++i) { int v = hist[i * NEXP + t]; hist[i * NEXP + t] = run; run += v; }
    tot1[t] = run;
  }
  __syncthreads();
#pragma unroll
  for (int e = 0; e < NEXP; ++e) h[e] = hist[t * NEXP + e];
#pragma unroll
  for (int i = 0; i < 8; ++i) {
    int e = le[i]; int pos = h[e]++;
    token_of[e * CAP + pos] = s0 + i;
    gate_of[e * CAP + pos]  = g1[s0 + i];
  }
  __syncthreads();

  // ---- pass 2: second-choice experts, offset by count1 ----
#pragma unroll
  for (int e = 0; e < NEXP; ++e) h[e] = 0;
#pragma unroll
  for (int i = 0; i < 8; ++i) { le[i] = e2[s0 + i]; h[le[i]]++; }
#pragma unroll
  for (int e = 0; e < NEXP; ++e) hist[t * NEXP + e] = h[e];
  __syncthreads();
  if (t < NEXP) {
    int run = tot1[t];
    for (int i = 0; i < 256; ++i) { int v = hist[i * NEXP + t]; hist[i * NEXP + t] = run; run += v; }
    total[t] = run;
  }
  __syncthreads();
#pragma unroll
  for (int e = 0; e < NEXP; ++e) h[e] = hist[t * NEXP + e];
#pragma unroll
  for (int i = 0; i < 8; ++i) {
    int e = le[i]; int pos = h[e]++;
    token_of[e * CAP + pos] = s0 + i;
    gate_of[e * CAP + pos]  = g2[s0 + i];
  }
}

// ---------------------------------------------------------------------------
// 4) fused expert FFN:  out[tok] += gate * ( relu(Xe @ w1) @ w2 )[row]
// grid = E * (CAP/32) blocks, 256 threads (8 waves). Block = 32 expert rows,
// so every B fragment feeds TWO wmma ops (row-halves) -> halves L2 weight
// traffic vs a 16-row tile. Chunked over DFF in 128-col slices:
// GEMM1 -> relu -> LDS bf16 -> GEMM2 accumulate. Dynamic LDS (~74KB/WGP).
// ---------------------------------------------------------------------------
__global__ __launch_bounds__(256) void ffn_kernel(const float* __restrict__ x,
                                                  const unsigned short* __restrict__ pw1_u,
                                                  const unsigned short* __restrict__ pw2_u,
                                                  const int* __restrict__ token_of,
                                                  const float* __restrict__ gate_of,
                                                  const int* __restrict__ total,
                                                  float* __restrict__ out) {
  extern __shared__ __align__(16) char smem[];
  __bf16* Xs    = (__bf16*)smem;
  __bf16* Hs    = (__bf16*)(smem + XS_BYTES);
  int*    tok_s = (int*)(smem + TOK_OFF);
  float*  gate_s = (float*)(smem + TOK_OFF + MROWS * 4);

  const __bf16* pw1 = (const __bf16*)pw1_u;
  const __bf16* pw2 = (const __bf16*)pw2_u;

  int e    = blockIdx.x >> 6;            // 64 row-tiles per expert
  int tile = blockIdx.x & 63;
  int row0 = tile * MROWS;
  int n    = total[e];
  if (row0 >= n) return;                 // uniform per block: no divergent barriers

  int tid = threadIdx.x;
  if (tid < MROWS) {
    int r = row0 + tid;
    int s = (r < n) ? token_of[e * CAP + r] : -1;
    tok_s[tid]  = s;
    gate_s[tid] = (r < n) ? gate_of[e * CAP + r] : 0.f;
  }
  __syncthreads();

  // gather 32 token rows -> LDS bf16 (rows beyond count are zeros)
  for (int idx = tid; idx < MROWS * DMODEL; idx += 256) {
    int r = idx >> 10, c = idx & (DMODEL - 1);
    int s = tok_s[r];
    float v = (s >= 0) ? x[(size_t)s * DMODEL + c] : 0.f;
    Xs[r * XS_STRIDE + c] = f2bf(v);
  }
  __syncthreads();

  int wave = tid >> 5;
  int lane = tid & 31;
  int hi   = lane >> 4;
  int ln   = lane & 15;

  v8f vzero;
#pragma unroll
  for (int i = 0; i < 8; ++i) vzero[i] = 0.f;

  v8f acc0[8], acc1[8];   // out rows 0..15 / 16..31, cols [wave*128, +128)
#pragma unroll
  for (int t = 0; t < 8; ++t) { acc0[t] = vzero; acc1[t] = vzero; }

  for (int ch = 0; ch < DFF_ / 128; ++ch) {
    // ---- GEMM1: this wave computes H[32 x 16] for cols ch*128 + wave*16 ----
    int ncolTile = ch * 8 + wave;        // w1 has 256 column tiles
    v8f h0 = vzero, h1 = vzero;
    for (int kt = 0; kt < DMODEL / 32; ++kt) {
      const __bf16* bp = pw1 + (size_t)(kt * 256 + ncolTile) * 512 + lane * 16;
      v16bf b  = load_b_frag(bp);
      v16bf aL = load_a_frag(Xs + ln * XS_STRIDE,        kt * 32, hi);
      v16bf aH = load_a_frag(Xs + (16 + ln) * XS_STRIDE, kt * 32, hi);
      h0 = __builtin_amdgcn_wmma_f32_16x16x32_bf16(false, aL, false, b, (short)0, h0, false, false);
      h1 = __builtin_amdgcn_wmma_f32_16x16x32_bf16(false, aH, false, b, (short)0, h1, false, false);
    }
    __syncthreads();   // previous chunk's GEMM2 done reading Hs
    // relu + bf16 -> Hs (C/D layout: VGPR i = row i + 8*hi, col = ln)
#pragma unroll
    for (int i = 0; i < 8; ++i) {
      float v0 = h0[i]; v0 = v0 > 0.f ? v0 : 0.f;
      float v1 = h1[i]; v1 = v1 > 0.f ? v1 : 0.f;
      Hs[(i + 8 * hi) * HS_STRIDE      + wave * 16 + ln] = f2bf(v0);
      Hs[(16 + i + 8 * hi) * HS_STRIDE + wave * 16 + ln] = f2bf(v1);
    }
    __syncthreads();
    // ---- GEMM2: acc += Hs[32 x 128] @ w2[chunk, wave*128 .. +128) ----
#pragma unroll
    for (int k2 = 0; k2 < 4; ++k2) {
      v16bf aL = load_a_frag(Hs + ln * HS_STRIDE,        k2 * 32, hi);
      v16bf aH = load_a_frag(Hs + (16 + ln) * HS_STRIDE, k2 * 32, hi);
#pragma unroll
      for (int nt = 0; nt < 8; ++nt) {
        const __bf16* bp = pw2 + (size_t)((ch * 4 + k2) * 64 + wave * 8 + nt) * 512 + lane * 16;
        v16bf b = load_b_frag(bp);
        acc0[nt] = __builtin_amdgcn_wmma_f32_16x16x32_bf16(false, aL, false, b, (short)0, acc0[nt], false, false);
        acc1[nt] = __builtin_amdgcn_wmma_f32_16x16x32_bf16(false, aH, false, b, (short)0, acc1[nt], false, false);
      }
    }
  }

  // ---- scatter-combine: out[token] += gate * acc (2 adds/elem, commutative) ----
#pragma unroll
  for (int nt = 0; nt < 8; ++nt) {
    int col = wave * 128 + nt * 16 + ln;
#pragma unroll
    for (int i = 0; i < 8; ++i) {
      int r0 = i + 8 * hi;
      int r1 = 16 + i + 8 * hi;
      int s0 = tok_s[r0];
      int s1 = tok_s[r1];
      if (s0 >= 0) atomicAdd(&out[(size_t)s0 * DMODEL + col], gate_s[r0] * acc0[nt][i]);
      if (s1 >= 0) atomicAdd(&out[(size_t)s1 * DMODEL + col], gate_s[r1] * acc1[nt][i]);
    }
  }
}

// ---------------------------------------------------------------------------
// launcher
// ---------------------------------------------------------------------------
extern "C" void kernel_launch(void* const* d_in, const int* in_sizes, int n_in,
                              void* d_out, int out_size, void* d_ws, size_t ws_size,
                              hipStream_t stream) {
  (void)in_sizes; (void)n_in; (void)out_size; (void)ws_size;
  const float* x  = (const float*)d_in[0];
  const float* Wg = (const float*)d_in[1];
  const float* w1 = (const float*)d_in[2];
  const float* w2 = (const float*)d_in[3];
  float* out = (float*)d_out;

  // workspace carve-up
  unsigned short* pw1 = (unsigned short*)d_ws;                 // 1024*4096 bf16
  unsigned short* pw2 = pw1 + (size_t)DMODEL * DFF_;           // 4096*1024 bf16
  int*   e1       = (int*)(pw2 + (size_t)DFF_ * DMODEL);
  int*   e2       = e1 + S_TOK;
  float* g1       = (float*)(e2 + S_TOK);
  float* g2       = g1 + S_TOK;
  int*   token_of = (int*)(g2 + S_TOK);                        // [E, CAP]
  float* gate_of  = (float*)(token_of + NEXP * CAP);           // [E, CAP]
  int*   total    = (int*)(gate_of + NEXP * CAP);              // [E]

  // 0) zero output (atomics accumulate onto it)
  zero_kernel<<<(S_TOK * DMODEL + 255) / 256, 256, 0, stream>>>(out, S_TOK * DMODEL);

  // 1) pack weights to bf16 fragment tiles (LDS-staged, coalesced both ways)
  pack_b_kernel<<<(DMODEL / 32) * (DFF_ / 256), 256, 0, stream>>>(w1, pw1, DFF_);
  pack_b_kernel<<<(DFF_ / 32) * (DMODEL / 256), 256, 0, stream>>>(w2, pw2, DMODEL);

  // 2) gating (one wave per token)
  gate_kernel<<<S_TOK / 8, 256, 0, stream>>>(x, Wg, e1, e2, g1, g2);

  // 3) parallel slot scan (single block)
  scan_kernel<<<1, 256, 0, stream>>>(e1, e2, g1, g2, token_of, gate_of, total);

  // 4) fused expert FFN + scatter-combine (dynamic LDS ~74KB)
  ffn_kernel<<<NEXP * (CAP / MROWS), 256, SMEM_BYTES, stream>>>(x, pw1, pw2, token_of,
                                                                gate_of, total, out);
}
